// TinyDecoderLM_91173565759651
// MI455X (gfx1250) — matrix operand order repaired and checked by reading, not compiled, not run on hardware
//
#include <hip/hip_runtime.h>
#include <cstddef>
#include <cstdint>

typedef unsigned short u16;
typedef __bf16 bf16_t;
typedef bf16_t v16bf __attribute__((ext_vector_type(16)));
typedef float v8f __attribute__((ext_vector_type(8)));

// Model constants (match reference)
#define MB   2          // batch
#define MS   1024       // seq
#define MD   1024       // d_model
#define MH   16         // heads
#define MHD  64         // head dim
#define ML   12         // layers
#define MF   4096       // mlp hidden
#define MV   32000      // vocab
#define MSCALE 0.125f   // 1/sqrt(64)

union FragBF {
    v16bf v;
    uint4 q[2];
    u16   u[16];
};

__device__ __forceinline__ u16 f2bf(float f) {
    union { float f; unsigned int u; } v;
    v.f = f;
    unsigned int r = v.u + 0x7FFFu + ((v.u >> 16) & 1u);
    return (u16)(r >> 16);
}
__device__ __forceinline__ float bf2f(u16 h) {
    union { unsigned int u; float f; } v;
    v.u = ((unsigned int)h) << 16;
    return v.f;
}

// ---- CDNA5 async copy global->LDS (ASYNCcnt-tracked, VGPR-bypassing) ----
__device__ __forceinline__ void async_b128(unsigned ldsOff, const void* gptr) {
    asm volatile("global_load_async_to_lds_b128 %0, %1, off"
                 :: "v"(ldsOff), "v"(gptr) : "memory");
}
__device__ __forceinline__ void wait_async0() {
    asm volatile("s_wait_asynccnt 0x0" ::: "memory");
}
__device__ __forceinline__ void wait_async4() {
    asm volatile("s_wait_asynccnt 0x4" ::: "memory");
}
__device__ __forceinline__ unsigned lds_off(const void* p) {
    return (unsigned)(uintptr_t)p;   // low 32 bits of generic addr = LDS offset
}

// ---------------------------------------------------------------------------
// Weight convert fp32 -> bf16 (no transpose): out[i] = bf16(in[i])
// ---------------------------------------------------------------------------
__global__ __launch_bounds__(256) void cvt_kernel(
    const float* __restrict__ in, u16* __restrict__ out)
{
    size_t i = (size_t)blockIdx.x * 1024 + threadIdx.x;
#pragma unroll
    for (int j = 0; j < 4; ++j) out[i + j * 256] = f2bf(in[i + j * 256]);
}

// ---------------------------------------------------------------------------
// Weight convert + transpose: in [K,N] fp32 -> out [N,K] bf16, 32x32 tiles
// ---------------------------------------------------------------------------
__global__ __launch_bounds__(256) void cvt_t_kernel(
    const float* __restrict__ in, u16* __restrict__ out, int K, int N)
{
    __shared__ float t[32][33];
    const int k0 = blockIdx.y * 32, n0 = blockIdx.x * 32;
    const int tx = threadIdx.x & 31, ty = threadIdx.x >> 5;
#pragma unroll
    for (int i = 0; i < 4; ++i)
        t[ty + 8 * i][tx] = in[(size_t)(k0 + ty + 8 * i) * N + n0 + tx];
    __syncthreads();
#pragma unroll
    for (int i = 0; i < 4; ++i)
        out[(size_t)(n0 + ty + 8 * i) * K + k0 + tx] = f2bf(t[tx][ty + 8 * i]);
}

// ---------------------------------------------------------------------------
// Embedding: x[b,s,:] = token_emb[idx[b,s],:] + pos_emb[s,:]   (fp32 residual)
// ---------------------------------------------------------------------------
__global__ __launch_bounds__(256) void embed_kernel(
    const int* __restrict__ idx, const float* __restrict__ tok,
    const float* __restrict__ pos, float* __restrict__ x)
{
    int row = blockIdx.x;          // b*S + s
    int s   = row & (MS - 1);
    int t   = idx[row];
    const float* tr = tok + (size_t)t * MD;
    const float* pr = pos + (size_t)s * MD;
    float* xr = x + (size_t)row * MD;
    for (int i = threadIdx.x; i < MD; i += 256)
        xr[i] = tr[i] + pr[i];
}

// ---------------------------------------------------------------------------
// LayerNorm over D=1024 (fp32 in, bf16 out), one block (256 thr) per row
// ---------------------------------------------------------------------------
__global__ __launch_bounds__(256) void ln_kernel(
    const float* __restrict__ x, const float* __restrict__ g,
    const float* __restrict__ bb, u16* __restrict__ y)
{
    __shared__ float red[256];
    const int tid = threadIdx.x;
    const float* xr = x + (size_t)blockIdx.x * MD;
    float lv[4];
    float s = 0.f;
#pragma unroll
    for (int i = 0; i < 4; ++i) { lv[i] = xr[tid + i * 256]; s += lv[i]; }
    red[tid] = s; __syncthreads();
    for (int off = 128; off > 0; off >>= 1) {
        if (tid < off) red[tid] += red[tid + off];
        __syncthreads();
    }
    float mean = red[0] * (1.f / MD);
    __syncthreads();
    float vs = 0.f;
#pragma unroll
    for (int i = 0; i < 4; ++i) { float d = lv[i] - mean; vs += d * d; }
    red[tid] = vs; __syncthreads();
    for (int off = 128; off > 0; off >>= 1) {
        if (tid < off) red[tid] += red[tid + off];
        __syncthreads();
    }
    float rstd = rsqrtf(red[0] * (1.f / MD) + 1e-5f);
    u16* yr = y + (size_t)blockIdx.x * MD;
#pragma unroll
    for (int i = 0; i < 4; ++i) {
        int c = tid + i * 256;
        yr[c] = f2bf((lv[i] - mean) * rstd * g[c] + bb[c]);
    }
}

// ---------------------------------------------------------------------------
// Tiled WMMA GEMM: out = act(A[M,K] * B^T + bias) (+ res)
//  - A bf16 [M,K] row-major, B bf16 [N,K] row-major (pre-transposed weights)
//  - double-buffered LDS, staged via GLOBAL_LOAD_ASYNC_TO_LDS_B128
//  - block 256 thr = 8 waves; tile 128x128x32; wave tile 64x32 (4x2 frags)
//  - optional fp32 output C and/or bf16 output C16
// ---------------------------------------------------------------------------
__global__ __launch_bounds__(256) void gemm_bf16_kernel(
    const u16* __restrict__ A, const u16* __restrict__ Bm,
    const float* __restrict__ bias, const float* __restrict__ res,
    float* __restrict__ C, u16* __restrict__ C16,
    int M, int N, int K, int actGelu)
{
    __shared__ u16 As[2][128 * 40];   // [m][k], stride 40 u16 (80B, 16B aligned)
    __shared__ u16 Bs[2][128 * 40];   // [n][k]

    const int m0 = blockIdx.y * 128;
    const int n0 = blockIdx.x * 128;
    const int tid   = threadIdx.x;
    const int lane  = tid & 31;
    const int wave  = tid >> 5;
    const int waveM = wave >> 2;   // 0..1
    const int waveN = wave & 3;    // 0..3

    const v8f vzero = {0.f,0.f,0.f,0.f,0.f,0.f,0.f,0.f};
    v8f acc[4][2];
#pragma unroll
    for (int i = 0; i < 4; ++i)
#pragma unroll
        for (int j = 0; j < 2; ++j) acc[i][j] = vzero;

    const int nK = K >> 5;

    // stage one 128x32 A tile + 128x32 B tile: 4 async b128 per thread
    auto stage = [&](int kt, int bsel) {
        const int k0 = kt << 5;
#pragma unroll
        for (int it = 0; it < 2; ++it) {
            int c = tid + it * 256;          // 0..511
            int row = c >> 2, cc = (c & 3) << 3;
            async_b128(lds_off(&As[bsel][row * 40 + cc]),
                       A + (size_t)(m0 + row) * K + k0 + cc);
        }
#pragma unroll
        for (int it = 0; it < 2; ++it) {
            int c = tid + it * 256;
            int row = c >> 2, cc = (c & 3) << 3;
            async_b128(lds_off(&Bs[bsel][row * 40 + cc]),
                       Bm + (size_t)(n0 + row) * K + k0 + cc);
        }
    };

    stage(0, 0);
    for (int kt = 0; kt < nK; ++kt) {
        const int cur = kt & 1;
        if (kt + 1 < nK) { stage(kt + 1, cur ^ 1); wait_async4(); }
        else             { wait_async0(); }
        __syncthreads();

        FragBF bf[2];
#pragma unroll
        for (int fn = 0; fn < 2; ++fn) {
            int n  = waveN * 32 + fn * 16 + (lane & 15);
            int kb = (lane >> 4) << 3;
            const uint4* p = (const uint4*)&Bs[cur][n * 40 + kb];
            bf[fn].q[0] = p[0];
            bf[fn].q[1] = p[2];
        }
#pragma unroll
        for (int fm = 0; fm < 4; ++fm) {
            FragBF af;
            int m  = waveM * 64 + fm * 16 + (lane & 15);
            int b0 = (lane >> 4) << 3;
            af.q[0] = *(const uint4*)&As[cur][m * 40 + b0];
            af.q[1] = *(const uint4*)&As[cur][m * 40 + b0 + 16];
#pragma unroll
            for (int fn = 0; fn < 2; ++fn)
                acc[fm][fn] = __builtin_amdgcn_wmma_f32_16x16x32_bf16(
                    false, af.v, false, bf[fn].v, (short)0, acc[fm][fn], false, false);
        }
        __syncthreads();
    }

    // ---- epilogue: bias, GELU(exact erf), residual, dual-precision store ----
#pragma unroll
    for (int fm = 0; fm < 4; ++fm) {
        int mbase = m0 + waveM * 64 + fm * 16 + ((lane >> 4) << 3);
#pragma unroll
        for (int fn = 0; fn < 2; ++fn) {
            int col = n0 + waveN * 32 + fn * 16 + (lane & 15);
            float bv = bias ? bias[col] : 0.f;
#pragma unroll
            for (int r = 0; r < 8; ++r) {
                float v = acc[fm][fn][r] + bv;
                if (actGelu) v = 0.5f * v * (1.f + erff(v * 0.70710678118f));
                size_t ci = (size_t)(mbase + r) * N + col;
                if (res) v += res[ci];
                if (C)   C[ci]   = v;
                if (C16) C16[ci] = f2bf(v);
            }
        }
    }
}

// ---------------------------------------------------------------------------
// Fused causal flash-attention + depth-memory, per (b,h, 64-query tile).
//  block = 128 thr = 4 waves; each wave owns 16 query rows. All inputs bf16.
//  QK^T and P*V via bf16 WMMA; online softmax; depth scores in epilogue.
// ---------------------------------------------------------------------------
__global__ __launch_bounds__(128) void attn_kernel(
    const u16* __restrict__ qr, const u16* __restrict__ kr,
    const u16* __restrict__ vr, const u16* __restrict__ qc,
    const u16* __restrict__ kcA, const u16* __restrict__ vcA,
    u16* __restrict__ ctx, int nPast)
{
    const int KS = 80;                       // u16 stride (160B rows)
    __shared__ u16  Kl[64 * 80];             // [key][d]  (B for QK^T)
    __shared__ u16  Vl[64 * 80];             // [d][key]  (B for P*V)
    __shared__ u16  Pl[4 * 16 * 80];         // per-wave P tile [m][key]
    __shared__ float Cl[64 * 68];            // ctx dump
    __shared__ float rowM[64], rowL[64];
    __shared__ float msl[64 * 12], wl[64 * 12];
    __shared__ float alphal[64], linvl[64];

    const int bh  = blockIdx.x;              // b*H + h
    const int b   = bh >> 4;
    const int hh  = bh & 15;
    const int qt  = blockIdx.y;
    const int tid  = threadIdx.x;
    const int lane = tid & 31;
    const int wave = tid >> 5;
    const size_t BSD = (size_t)MB * MS * MD;

    auto gidx = [&](int sRow, int d) -> size_t {
        return ((size_t)(b * MS + sRow)) * MD + (size_t)hh * MHD + d;
    };

    // ---- load Q fragments (raw bf16; 1/sqrt(hd) applied post-WMMA) ----
    FragBF qf[2];
    {
        int sq = qt * 64 + wave * 16 + (lane & 15);
        int b0 = (lane >> 4) << 3;
#pragma unroll
        for (int kk = 0; kk < 2; ++kk) {
            qf[kk].q[0] = *(const uint4*)(qr + gidx(sq, kk * 32 + b0));
            qf[kk].q[1] = *(const uint4*)(qr + gidx(sq, kk * 32 + b0 + 16));
        }
    }

    float mrun[8], lrun[8];
    v8f accC[4];
    const v8f vzero = {0.f,0.f,0.f,0.f,0.f,0.f,0.f,0.f};
#pragma unroll
    for (int r = 0; r < 8; ++r) { mrun[r] = -1e30f; lrun[r] = 0.f; }
#pragma unroll
    for (int i = 0; i < 4; ++i) accC[i] = vzero;

    for (int jt = 0; jt <= qt; ++jt) {
        __syncthreads();
        // ---- K tile (64x64, [key][d]) via async copies; V transposed by hand ----
#pragma unroll
        for (int it = 0; it < 4; ++it) {
            int c = tid + it * 128;              // 0..511
            int key = c >> 3, cc = (c & 7) << 3;
            async_b128(lds_off(&Kl[key * KS + cc]),
                       kr + gidx(jt * 64 + key, cc));
        }
        {
            int key = tid >> 1;
            int dh  = (tid & 1) << 5;            // 0 or 32
            const u16* vs = vr + gidx(jt * 64 + key, dh);
#pragma unroll
            for (int i = 0; i < 32; ++i) Vl[(dh + i) * KS + key] = vs[i];
        }
        wait_async0();
        __syncthreads();

        // ---- scores: S(16x64) = Q(16x64d) * K^T, then *SCALE ----
        v8f sc[4];
#pragma unroll
        for (int fn = 0; fn < 4; ++fn) {
            sc[fn] = vzero;
#pragma unroll
            for (int kk = 0; kk < 2; ++kk) {
                FragBF kf;
                int n  = fn * 16 + (lane & 15);                 // key col
                int kb = kk * 32 + ((lane >> 4) << 3);          // d offset
                const uint4* p = (const uint4*)&Kl[n * KS + kb];
                kf.q[0] = p[0]; kf.q[1] = p[2];
                sc[fn] = __builtin_amdgcn_wmma_f32_16x16x32_bf16(
                    false, qf[kk].v, false, kf.v, (short)0, sc[fn], false, false);
            }
            sc[fn] = sc[fn] * MSCALE;
        }

        // ---- causal mask on diagonal tile ----
        if (jt == qt) {
            int rbase = wave * 16 + ((lane >> 4) << 3);
#pragma unroll
            for (int fn = 0; fn < 4; ++fn) {
                int lk = fn * 16 + (lane & 15);
#pragma unroll
                for (int r = 0; r < 8; ++r)
                    if (lk > rbase + r) sc[fn][r] = -1e30f;
            }
        }

        // ---- online softmax update ----
#pragma unroll
        for (int r = 0; r < 8; ++r) {
            float v = fmaxf(fmaxf(sc[0][r], sc[1][r]), fmaxf(sc[2][r], sc[3][r]));
            v = fmaxf(v, __shfl_xor(v, 8, 32));
            v = fmaxf(v, __shfl_xor(v, 4, 32));
            v = fmaxf(v, __shfl_xor(v, 2, 32));
            v = fmaxf(v, __shfl_xor(v, 1, 32));
            float mnew = fmaxf(mrun[r], v);
            float al = __expf(mrun[r] - mnew);
            mrun[r] = mnew;
            lrun[r] *= al;
#pragma unroll
            for (int fn = 0; fn < 4; ++fn) accC[fn][r] *= al;
        }

        // ---- p = exp(s - m), write P tile (C-layout -> LDS), row sums ----
        {
            u16* pw = &Pl[wave * 16 * KS];
#pragma unroll
            for (int r = 0; r < 8; ++r) {
                int mloc = ((lane >> 4) << 3) + r;
                float rs = 0.f;
#pragma unroll
                for (int fn = 0; fn < 4; ++fn) {
                    float p = __expf(sc[fn][r] - mrun[r]);
                    rs += p;
                    pw[mloc * KS + fn * 16 + (lane & 15)] = f2bf(p);
                }
                rs += __shfl_xor(rs, 8, 32);
                rs += __shfl_xor(rs, 4, 32);
                rs += __shfl_xor(rs, 2, 32);
                rs += __shfl_xor(rs, 1, 32);
                lrun[r] += rs;
            }
        }
        __syncthreads();

        // ---- ctx += P(16x64) * V(64x64) ----
#pragma unroll
        for (int kk = 0; kk < 2; ++kk) {
            FragBF pf;
            int m  = lane & 15;
            int b0 = (lane >> 4) << 3;
            const u16* pr = &Pl[wave * 16 * KS + m * KS + kk * 32];
            pf.q[0] = *(const uint4*)(pr + b0);
            pf.q[1] = *(const uint4*)(pr + b0 + 16);
#pragma unroll
            for (int fn = 0; fn < 4; ++fn) {
                FragBF vf;
                int n  = fn * 16 + (lane & 15);                // d col
                int kb = kk * 32 + ((lane >> 4) << 3);         // key offset
                const uint4* vp = (const uint4*)&Vl[n * KS + kb];
                vf.q[0] = vp[0]; vf.q[1] = vp[2];
                accC[fn] = __builtin_amdgcn_wmma_f32_16x16x32_bf16(
                    false, pf.v, false, vf.v, (short)0, accC[fn], false, false);
            }
        }
    } // jt

    // ---- dump ctx + stats ----
    __syncthreads();
    {
        int rb = wave * 16 + ((lane >> 4) << 3);
#pragma unroll
        for (int fn = 0; fn < 4; ++fn)
#pragma unroll
            for (int r = 0; r < 8; ++r)
                Cl[(rb + r) * 68 + fn * 16 + (lane & 15)] = accC[fn][r];
        if ((lane & 15) == 0) {
#pragma unroll
            for (int r = 0; r < 8; ++r) { rowM[rb + r] = mrun[r]; rowL[rb + r] = lrun[r]; }
        }
    }
    __syncthreads();

    if (nPast > 0) {
        // depth scores vs past shared-KV layers
        for (int job = tid; job < 64 * nPast; job += 128) {
            int row = job / nPast, p = job % nPast;
            int sq = qt * 64 + row;
            const u16* qv = qc + gidx(sq, 0);
            const u16* kv = kcA + (size_t)p * BSD + gidx(sq, 0);
            float dot = 0.f;
#pragma unroll 8
            for (int d = 0; d < MHD; ++d) dot += bf2f(qv[d]) * bf2f(kv[d]);
            msl[row * 12 + p] = dot * MSCALE;
        }
        __syncthreads();
        for (int row = tid; row < 64; row += 128) {
            float m0 = rowM[row], l0 = rowL[row];
            float mf = m0;
            for (int p = 0; p < nPast; ++p) mf = fmaxf(mf, msl[row * 12 + p]);
            float al = __expf(m0 - mf);
            float lf = l0 * al;
            for (int p = 0; p < nPast; ++p) {
                float w = __expf(msl[row * 12 + p] - mf);
                wl[row * 12 + p] = w;
                lf += w;
            }
            alphal[row] = al;
            linvl[row] = 1.f / lf;
        }
        __syncthreads();
        for (int e = tid; e < 64 * 64; e += 128) {
            int row = e >> 6, d = e & 63;
            int sq = qt * 64 + row;
            float v = Cl[row * 68 + d] * alphal[row];
            for (int p = 0; p < nPast; ++p)
                v += wl[row * 12 + p] * bf2f(vcA[(size_t)p * BSD + gidx(sq, d)]);
            ctx[gidx(sq, d)] = f2bf(v * linvl[row]);
        }
    } else {
        for (int e = tid; e < 64 * 64; e += 128) {
            int row = e >> 6, d = e & 63;
            int sq = qt * 64 + row;
            ctx[gidx(sq, d)] = f2bf(Cl[row * 68 + d] / rowL[row]);
        }
    }
}

// ---------------------------------------------------------------------------
// Host orchestration
// ---------------------------------------------------------------------------
extern "C" void kernel_launch(void* const* d_in, const int* in_sizes, int n_in,
                              void* d_out, int out_size, void* d_ws, size_t ws_size,
                              hipStream_t stream) {
    (void)in_sizes; (void)n_in; (void)out_size; (void)ws_size;

    const int*   idx     = (const int*)  d_in[0];
    const float* tok     = (const float*)d_in[1];
    const float* pos     = (const float*)d_in[2];
    const float* Wq_row  = (const float*)d_in[3];
    const float* bq_row  = (const float*)d_in[4];
    const float* Wk_row  = (const float*)d_in[5];
    const float* bk_row  = (const float*)d_in[6];
    const float* Wv_row  = (const float*)d_in[7];
    const float* bv_row  = (const float*)d_in[8];
    const float* Wq_col  = (const float*)d_in[9];
    const float* bq_col  = (const float*)d_in[10];
    const float* Wout    = (const float*)d_in[11];
    const float* bout    = (const float*)d_in[12];
    const float* Wk_sh   = (const float*)d_in[13];
    const float* bk_sh   = (const float*)d_in[14];
    const float* Wv_sh   = (const float*)d_in[15];
    const float* bv_sh   = (const float*)d_in[16];
    const float* ln1_g   = (const float*)d_in[17];
    const float* ln1_b   = (const float*)d_in[18];
    const float* ln2_g   = (const float*)d_in[19];
    const float* ln2_b   = (const float*)d_in[20];
    const float* W1      = (const float*)d_in[21];
    const float* b1      = (const float*)d_in[22];
    const float* W2      = (const float*)d_in[23];
    const float* b2      = (const float*)d_in[24];
    const float* lnf_g   = (const float*)d_in[25];
    const float* lnf_b   = (const float*)d_in[26];

    const size_t BSD = (size_t)MB * MS * MD;
    const size_t DD  = (size_t)MD * MD;
    const size_t DF  = (size_t)MD * MF;
    const int    T   = MB * MS;

    // ---- workspace carve-out ----
    char* wp = (char*)d_ws;
    auto carve = [&](size_t bytes) -> void* {
        void* p = (void*)wp;
        wp += (bytes + 255) & ~(size_t)255;
        return p;
    };
    float* x     = (float*)carve(BSD * 4);            // fp32 residual stream
    u16* h16     = (u16*)carve(BSD * 2);
    u16* h216    = (u16*)carve(BSD * 2);
    u16* qr16    = (u16*)carve(BSD * 2);
    u16* kr16    = (u16*)carve(BSD * 2);
    u16* vr16    = (u16*)carve(BSD * 2);
    u16* qc16    = (u16*)carve(BSD * 2);
    u16* ctx16   = (u16*)carve(BSD * 2);
    u16* mlp16   = (u16*)carve((size_t)T * MF * 2);
    u16* kcA16   = (u16*)carve((size_t)ML * BSD * 2);
    u16* vcA16   = (u16*)carve((size_t)ML * BSD * 2);
    u16* Wt      = (u16*)carve((size_t)ML * 5 * DD * 2);  // per-layer: q,k,v,qc,out
    u16* W1t     = (u16*)carve((size_t)ML * DF * 2);      // [F,D]
    u16* W2t     = (u16*)carve((size_t)ML * DF * 2);      // [D,F]
    u16* Wksh_t  = (u16*)carve(DD * 2);
    u16* Wvsh_t  = (u16*)carve(DD * 2);
    u16* tok16   = (u16*)carve((size_t)MV * MD * 2);

    dim3 blk256(256), blk128(128);
    dim3 gDDt(MD / 32, MD / 32);
    dim3 gDFt(MF / 32, MD / 32);   // W1 [D,F] -> [F,D]
    dim3 gFDt(MD / 32, MF / 32);   // W2 [F,D] -> [D,F]
    dim3 gDD(MD / 128, T / 128);
    dim3 gDF(MF / 128, T / 128);
    dim3 gV (MV / 128, T / 128);

    // ---- one-time (per launch) weight conversion to bf16, pre-transposed ----
    for (int l = 0; l < ML; ++l) {
        cvt_t_kernel<<<gDDt, blk256, 0, stream>>>(Wq_row + l * DD, Wt + (size_t)(l * 5 + 0) * DD, MD, MD);
        cvt_t_kernel<<<gDDt, blk256, 0, stream>>>(Wk_row + l * DD, Wt + (size_t)(l * 5 + 1) * DD, MD, MD);
        cvt_t_kernel<<<gDDt, blk256, 0, stream>>>(Wv_row + l * DD, Wt + (size_t)(l * 5 + 2) * DD, MD, MD);
        cvt_t_kernel<<<gDDt, blk256, 0, stream>>>(Wq_col + l * DD, Wt + (size_t)(l * 5 + 3) * DD, MD, MD);
        cvt_t_kernel<<<gDDt, blk256, 0, stream>>>(Wout   + l * DD, Wt + (size_t)(l * 5 + 4) * DD, MD, MD);
        cvt_t_kernel<<<gDFt, blk256, 0, stream>>>(W1 + l * DF, W1t + l * DF, MD, MF);
        cvt_t_kernel<<<gFDt, blk256, 0, stream>>>(W2 + l * DF, W2t + l * DF, MF, MD);
    }
    cvt_t_kernel<<<gDDt, blk256, 0, stream>>>(Wk_sh, Wksh_t, MD, MD);
    cvt_t_kernel<<<gDDt, blk256, 0, stream>>>(Wv_sh, Wvsh_t, MD, MD);
    cvt_kernel<<<(MV * MD) / 1024, blk256, 0, stream>>>(tok, tok16);   // [V,D]=[N,K]

    // ---- forward ----
    embed_kernel<<<T, blk256, 0, stream>>>(idx, tok, pos, x);

    for (int l = 0; l < ML; ++l) {
        const size_t oD = (size_t)l * MD;
        const u16* Wl = Wt + (size_t)l * 5 * DD;

        ln_kernel<<<T, blk256, 0, stream>>>(x, ln1_g + oD, ln1_b + oD, h16);

        gemm_bf16_kernel<<<gDD, blk256, 0, stream>>>(h16, Wl + 0 * DD, bq_row + oD, nullptr, nullptr, qr16, T, MD, MD, 0);
        gemm_bf16_kernel<<<gDD, blk256, 0, stream>>>(h16, Wl + 1 * DD, bk_row + oD, nullptr, nullptr, kr16, T, MD, MD, 0);
        gemm_bf16_kernel<<<gDD, blk256, 0, stream>>>(h16, Wl + 2 * DD, bv_row + oD, nullptr, nullptr, vr16, T, MD, MD, 0);
        gemm_bf16_kernel<<<gDD, blk256, 0, stream>>>(h16, Wl + 3 * DD, bq_col + oD, nullptr, nullptr, qc16, T, MD, MD, 0);
        gemm_bf16_kernel<<<gDD, blk256, 0, stream>>>(h16, Wksh_t, bk_sh, nullptr, nullptr, kcA16 + (size_t)l * BSD, T, MD, MD, 0);
        gemm_bf16_kernel<<<gDD, blk256, 0, stream>>>(h16, Wvsh_t, bv_sh, nullptr, nullptr, vcA16 + (size_t)l * BSD, T, MD, MD, 0);

        attn_kernel<<<dim3(MB * MH, MS / 64), blk128, 0, stream>>>(
            qr16, kr16, vr16, qc16, kcA16, vcA16, ctx16, l);

        // x = x + ctx @ Wout + bout
        gemm_bf16_kernel<<<gDD, blk256, 0, stream>>>(ctx16, Wl + 4 * DD, bout + oD, x, x, nullptr, T, MD, MD, 0);

        ln_kernel<<<T, blk256, 0, stream>>>(x, ln2_g + oD, ln2_b + oD, h216);

        // mlp = gelu(h2 @ W1 + b1)   (bf16-only output)
        gemm_bf16_kernel<<<gDF, blk256, 0, stream>>>(h216, W1t + (size_t)l * DF, b1 + (size_t)l * MF,
                                                     nullptr, nullptr, mlp16, T, MF, MD, 1);
        // x = x + mlp @ W2 + b2
        gemm_bf16_kernel<<<gDD, blk256, 0, stream>>>(mlp16, W2t + (size_t)l * DF, b2 + oD,
                                                     x, x, nullptr, T, MD, MF, 0);
    }

    // final LN + tied lm head (token_emb already [V,D]=[N,K] bf16)
    ln_kernel<<<T, blk256, 0, stream>>>(x, lnf_g, lnf_b, h16);
    gemm_bf16_kernel<<<gV, blk256, 0, stream>>>(h16, tok16, nullptr, nullptr,
                                                (float*)d_out, nullptr, T, MV, MD, 0);
}
